// ContrastiveLoss_56822417326450
// MI455X (gfx1250) — matrix operand order repaired and checked by reading, not compiled, not run on hardware
//
#include <hip/hip_runtime.h>
#include <hip/hip_bf16.h>
#include <math.h>

// ---------------------------------------------------------------------------
// Contrastive (InfoNCE) loss, one direction, for 16384x512 fp32 features.
//   f1n = normalize(f1); f2n = normalize(f2)
//   logits = f1n @ f2n^T / 0.07      (never materialized)
//   loss = mean_i( log(sum_j exp(logits[i,j])) - logits[i,i] )
// Compute-bound (275 GFLOP on 64MB) -> bf16 WMMA GEMM fused with row sum-exp.
// A fragments live in VGPRs for the whole column sweep; B operands are
// streamed into double-buffered LDS by the Tensor Data Mover and consumed via
// a software-pipelined ds_load stream, so the matrix pipe never waits on VMEM
// and LDS latency is hidden behind the WMMA chains.
// ---------------------------------------------------------------------------

#define DIM        512
#define INV_T      14.285714285714286f           // 1/0.07
#define LOG2E      1.4426950408889634f
#define LN2        0.6931471805599453f
#define EPS_NRM    1e-12f

typedef __attribute__((ext_vector_type(16))) __bf16       v16bf;
typedef __attribute__((ext_vector_type(8)))  float        v8f;
typedef __attribute__((ext_vector_type(4)))  unsigned int u32x4;
typedef __attribute__((ext_vector_type(8)))  int          i32x8;
typedef __attribute__((ext_vector_type(4)))  int          i32x4;

#if defined(__has_builtin)
#if __has_builtin(__builtin_amdgcn_tensor_load_to_lds) && \
    __has_builtin(__builtin_amdgcn_s_wait_tensorcnt)
#define CL_HAS_TDM 1
#endif
#endif
#ifndef CL_HAS_TDM
#define CL_HAS_TDM 0
#endif

__device__ __forceinline__ v16bf make_v16bf(u32x4 lo, u32x4 hi) {
    union { u32x4 u[2]; v16bf v; } c;
    c.u[0] = lo; c.u[1] = hi;
    return c.v;
}

// load one 32B B fragment (contiguous 16 bf16) from LDS row `row`, offset `bo`
__device__ __forceinline__ v16bf ldB(const __bf16* Bs, int row, int bo) {
    const __bf16* p = Bs + (size_t)row * DIM + bo;
    return make_v16bf(*(const u32x4*)p, *(const u32x4*)(p + 8));
}

// round-to-nearest-even fp32 -> bf16, packed pair into one dword
__device__ __forceinline__ unsigned pack2_bf16(float lo, float hi) {
    unsigned a = __float_as_uint(lo);
    unsigned b = __float_as_uint(hi);
    a = (a + 0x7FFFu + ((a >> 16) & 1u)) >> 16;
    b = (b + 0x7FFFu + ((b >> 16) & 1u)) >> 16;
    return a | (b << 16);
}

#if CL_HAS_TDM
// ---------------------------------------------------------------------------
// Issue one TDM 2-D tile load: `rows` consecutive 512-element bf16 rows from
// global `gp` into LDS byte address `lds_addr`.  D# per ISA 8.3/8.4:
//   group0: count=1 | lds_addr | global_addr[56:0] | type=2
//   group1: data_size=1(2B), tensor_dim0=512, tensor_dim1=16384,
//           tile_dim0=512, tile_dim1=rows, tensor_dim0_stride=512
// Tracked with TENSORcnt.  EXEC is ignored; call from one wave only.
// ---------------------------------------------------------------------------
__device__ __forceinline__ void tdm_load_rows(unsigned lds_addr,
                                              const void* gp, int rows) {
    unsigned long long ga = (unsigned long long)(uintptr_t)gp;
    u32x4 g0 = {};
    g0[0] = 1u;                                          // count=1 (valid)
    g0[1] = lds_addr;                                    // LDS byte address
    g0[2] = (unsigned)ga;                                // global_addr[31:0]
    g0[3] = (unsigned)((ga >> 32) & 0x01FFFFFFu)         // global_addr[56:32]
          | (2u << 30);                                  // type=2 ("image")
    i32x8 g1 = {};
    g1[0] = (int)(1u << 16);                             // data_size=1 -> 2B
    g1[1] = (int)(512u << 16);                           // tensor_dim0 = 512
    g1[2] = (int)(16384u << 16);                         // tensor_dim1 = 16384
    g1[3] = (int)(512u << 16);                           // tile_dim0 = 512
    g1[4] = rows;                                        // tile_dim1 = rows
    g1[5] = 512;                                         // tensor_dim0_stride
    i32x4 z = {};
#if __clang_major__ >= 23
    i32x8 z8 = {};
    __builtin_amdgcn_tensor_load_to_lds(g0, g1, z, z, z8, 0);
#else
    __builtin_amdgcn_tensor_load_to_lds(g0, g1, z, z, 0);
#endif
}
#endif // CL_HAS_TDM

// ---------------------------------------------------------------------------
// Kernel 1: per-row L2 norms of f1 and f2, exact fp32 diagonal logit,
//           bf16 normalized copies written to workspace.
// ---------------------------------------------------------------------------
__global__ __launch_bounds__(256)
void cl_normalize_kernel(const float* __restrict__ f1,
                         const float* __restrict__ f2,
                         unsigned* __restrict__ f1n,   // [N, 256] packed bf16x2
                         unsigned* __restrict__ f2n,
                         float* __restrict__ diag)
{
    const int row  = blockIdx.x;
    const int tid  = threadIdx.x;
    const int lane = tid & 31;
    const int wv   = tid >> 5;

    const float2* r1 = (const float2*)(f1 + (size_t)row * DIM);
    const float2* r2 = (const float2*)(f2 + (size_t)row * DIM);
    float2 x1 = r1[tid];
    float2 x2 = r2[tid];

    float s1  = x1.x * x1.x + x1.y * x1.y;
    float s2  = x2.x * x2.x + x2.y * x2.y;
    float s12 = x1.x * x2.x + x1.y * x2.y;

    #pragma unroll
    for (int off = 16; off > 0; off >>= 1) {
        s1  += __shfl_xor(s1,  off, 32);
        s2  += __shfl_xor(s2,  off, 32);
        s12 += __shfl_xor(s12, off, 32);
    }

    __shared__ float red1[8], red2[8], red12[8];
    __shared__ float bc[2];
    if (lane == 0) { red1[wv] = s1; red2[wv] = s2; red12[wv] = s12; }
    __syncthreads();
    if (tid == 0) {
        float a = 0.f, b = 0.f, c = 0.f;
        #pragma unroll
        for (int i = 0; i < 8; ++i) { a += red1[i]; b += red2[i]; c += red12[i]; }
        float i1 = 1.0f / fmaxf(sqrtf(a), EPS_NRM);
        float i2 = 1.0f / fmaxf(sqrtf(b), EPS_NRM);
        bc[0] = i1; bc[1] = i2;
        diag[row] = c * i1 * i2 * INV_T;   // exact fp32 diagonal logit
    }
    __syncthreads();
    const float i1 = bc[0], i2 = bc[1];

    f1n[(size_t)row * 256 + tid] = pack2_bf16(x1.x * i1, x1.y * i1);
    f2n[(size_t)row * 256 + tid] = pack2_bf16(x2.x * i2, x2.y * i2);
}

// ---------------------------------------------------------------------------
// Kernel 2: fused bf16 WMMA GEMM strip + row sum-of-exp.
// Block = 128 threads (4 waves); block owns 64 A-rows.  A is TDM-staged into
// LDS once, then each wave pins its 16 A fragments (128 VGPRs) in registers
// for the entire column sweep.  B stages (32 rows, 32 KB) are double-buffered
// in LDS and filled by the Tensor Data Mover overlapped with the 32 WMMAs of
// the current stage; inside a stage the B ds_loads are software-pipelined one
// K-step ahead of the WMMAs.  |logits| <= 1/0.07 so plain sum-exp is
// fp32-safe (no running max).
// LDS map (dynamic, 128 KB):  [0,64K) A   [64K,96K) B buf0   [96K,128K) B buf1
// ---------------------------------------------------------------------------
extern __shared__ unsigned char cl_smem[];

__global__ __launch_bounds__(128)
void cl_gemm_lse_kernel(const unsigned short* __restrict__ f1n,  // bf16 bits
                        const unsigned short* __restrict__ f2n,
                        float* __restrict__ lse,
                        int N)
{
    const int tid  = threadIdx.x;
    const int lane = tid & 31;
    const int wv   = tid >> 5;
    const int half = lane >> 4;                  // 0: lanes 0-15, 1: 16-31
    const int l15  = lane & 15;
    const int rowBlock = blockIdx.x * 64;
    const int nTiles = N >> 4;                   // 16-column tiles

    unsigned char* smA  = cl_smem;               // 64 KB A strip
    unsigned char* smB0 = cl_smem + 65536;       // 32 KB B stage buffer 0
    unsigned char* smB1 = cl_smem + 65536 + 32768;

    // ---- stage A strip (64 rows) and first B stage (32 rows) --------------
#if CL_HAS_TDM
    if (wv == 0) {
        tdm_load_rows((unsigned)(uintptr_t)smA,  f1n + (size_t)rowBlock * DIM, 64);
        tdm_load_rows((unsigned)(uintptr_t)smB0, f2n, 32);
        __builtin_amdgcn_s_wait_tensorcnt(0);
    }
#else
    {
        const u32x4* gA = (const u32x4*)(f1n + (size_t)rowBlock * DIM);
        u32x4* dA = (u32x4*)smA;
        #pragma unroll 4
        for (int i = tid; i < 4096; i += 128) dA[i] = gA[i];
        const u32x4* gB = (const u32x4*)f2n;
        u32x4* dB = (u32x4*)smB0;
        #pragma unroll 4
        for (int i = tid; i < 2048; i += 128) dB[i] = gB[i];
    }
#endif
    __syncthreads();

    // ---- pin this wave's 16 A fragments in registers for the whole sweep --
    // A 16x32 bf16 fragment: lane row l15, contiguous 32B chunk at
    // k = kk + half*16 (matches B).
    const __bf16* As = (const __bf16*)smA;
    const int aRow = wv * 16 + l15;
    v16bf afrag[16];
    #pragma unroll
    for (int kk = 0; kk < DIM; kk += 32) {
        const __bf16* ap = As + (size_t)aRow * DIM + kk + half * 16;
        afrag[kk >> 5] = make_v16bf(*(const u32x4*)ap, *(const u32x4*)(ap + 8));
    }

    float s[8];
    #pragma unroll
    for (int e = 0; e < 8; ++e) s[e] = 0.0f;

    const float CEXP = INV_T * LOG2E;            // exp(x/T) = exp2(x * CEXP)
    int cur = 0;

    for (int jt = 0; jt < nTiles; jt += 2) {
        // ---- kick off the TDM for the NEXT 32-column stage (overlapped) ----
#if CL_HAS_TDM
        if (wv == 0 && jt + 2 < nTiles) {
            unsigned char* nxt = cur ? smB0 : smB1;
            tdm_load_rows((unsigned)(uintptr_t)nxt,
                          f2n + (size_t)(jt + 2) * 16 * DIM, 32);
        }
#endif
        // pre-warm L2 two stages ahead of the TDM stream
        {
            int pr = (jt << 4) + 64 + lane;
            if (pr < N) __builtin_prefetch(f2n + (size_t)pr * DIM, 0, 0);
        }

        const __bf16* Bs = (const __bf16*)(cur ? smB1 : smB0);
        v8f acc0 = {};                            // two independent WMMA chains
        v8f acc1 = {};

        // B 32x16 bf16 fragment: lane column l15, contiguous 32B chunk at
        // k = kk + half*16 of rows l15 / 16+l15.  Software-pipelined one
        // K-step ahead so the ds_loads hide behind two WMMA issues.
        v16bf bc0 = ldB(Bs, l15,      half * 16);
        v16bf bc1 = ldB(Bs, 16 + l15, half * 16);
        #pragma unroll
        for (int kk = 0; kk < DIM; kk += 32) {
            v16bf bn0 = bc0, bn1 = bc1;
            if (kk + 32 < DIM) {
                const int bo = kk + 32 + half * 16;
                bn0 = ldB(Bs, l15,      bo);
                bn1 = ldB(Bs, 16 + l15, bo);
            }
            acc0 = __builtin_amdgcn_wmma_f32_16x16x32_bf16(
                       false, afrag[kk >> 5], false, bc0, (short)0, acc0,
                       false, false);
            acc1 = __builtin_amdgcn_wmma_f32_16x16x32_bf16(
                       false, afrag[kk >> 5], false, bc1, (short)0, acc1,
                       false, false);
            bc0 = bn0; bc1 = bn1;
        }

        // C/D layout: lane holds column l15, rows (half*8 + e); fold the
        // lane's 16 logits into per-row running sum-of-exp.
        #pragma unroll
        for (int e = 0; e < 8; ++e) {
            s[e] += __builtin_amdgcn_exp2f(acc0[e] * CEXP)
                  + __builtin_amdgcn_exp2f(acc1[e] * CEXP);
        }

        // ---- fence the stage swap: my TDM done, everyone past the reads ----
#if CL_HAS_TDM
        if (wv == 0) __builtin_amdgcn_s_wait_tensorcnt(0);
#else
        __syncthreads();
        if (jt + 2 < nTiles) {
            unsigned char* nxt = cur ? smB0 : smB1;
            const u32x4* gB = (const u32x4*)(f2n + (size_t)(jt + 2) * 16 * DIM);
            u32x4* dB = (u32x4*)nxt;
            #pragma unroll 4
            for (int i = tid; i < 2048; i += 128) dB[i] = gB[i];
        }
#endif
        __syncthreads();
        cur ^= 1;
    }

    // reduce over the 16 column-lanes of each half (xor 1,2,4,8 keeps bit4)
    #pragma unroll
    for (int e = 0; e < 8; ++e) {
        float v = s[e];
        v += __shfl_xor(v, 1, 32);
        v += __shfl_xor(v, 2, 32);
        v += __shfl_xor(v, 4, 32);
        v += __shfl_xor(v, 8, 32);
        s[e] = v;
    }

    if (l15 == 0) {
        const int rbase = rowBlock + wv * 16 + half * 8;
        #pragma unroll
        for (int e = 0; e < 8; ++e)
            lse[rbase + e] = __builtin_amdgcn_logf(s[e]) * LN2;
    }
}

// ---------------------------------------------------------------------------
// Kernel 3: loss = mean(lse - diag)
// ---------------------------------------------------------------------------
__global__ __launch_bounds__(256)
void cl_loss_kernel(const float* __restrict__ lse,
                    const float* __restrict__ diag,
                    float* __restrict__ out, int N)
{
    __shared__ float red[256];
    const int tid = threadIdx.x;
    float a = 0.0f;
    for (int i = tid; i < N; i += 256) a += lse[i] - diag[i];
    red[tid] = a;
    __syncthreads();
    #pragma unroll
    for (int stg = 128; stg > 0; stg >>= 1) {
        if (tid < stg) red[tid] += red[tid + stg];
        __syncthreads();
    }
    if (tid == 0) out[0] = red[0] / (float)N;
}

// ---------------------------------------------------------------------------
extern "C" void kernel_launch(void* const* d_in, const int* in_sizes, int n_in,
                              void* d_out, int out_size, void* d_ws, size_t ws_size,
                              hipStream_t stream)
{
    const float* f1 = (const float*)d_in[0];
    const float* f2 = (const float*)d_in[1];
    const int N = in_sizes[0] / DIM;             // 16384

    // workspace layout: f1n | f2n (bf16, packed) | diag | lse
    unsigned* f1n  = (unsigned*)d_ws;                          // N*256 dwords
    unsigned* f2n  = f1n + (size_t)N * 256;                    // N*256 dwords
    float*    diag = (float*)(f2n + (size_t)N * 256);          // N floats
    float*    lse  = diag + N;                                 // N floats

    cl_normalize_kernel<<<N, 256, 0, stream>>>(f1, f2, f1n, f2n, diag);

    cl_gemm_lse_kernel<<<N / 64, 128, 131072, stream>>>(
        (const unsigned short*)f1n, (const unsigned short*)f2n, lse, N);

    cl_loss_kernel<<<1, 256, 0, stream>>>(lse, diag, (float*)d_out, N);
}
